// TransformerEncoder_11141145166489
// MI455X (gfx1250) — matrix operand order repaired and checked by reading, not compiled, not run on hardware
//
#include <hip/hip_runtime.h>

typedef unsigned short u16;
typedef __attribute__((ext_vector_type(8)))  __bf16 bf8v;
typedef __attribute__((ext_vector_type(16))) __bf16 bf16v;
typedef __attribute__((ext_vector_type(8)))  float  f8v;

#define DEVI __device__ __forceinline__

constexpr int NTOK   = 3136;   // 56*56 tokens
constexpr int CDIM   = 256;
constexpr int NHD    = 8;
constexpr int DH     = 32;
constexpr int DFF_   = 1024;
constexpr int LAYERS = 4;
constexpr int HH     = 56, WWI = 56;
constexpr int TOPK_  = 16;
constexpr int USEN   = 10;
constexpr int NQC    = 1568;   // gram row chunk (2 chunks, 32-divisible)
constexpr int CMID   = 128;

// ---------- bf16 helpers ----------
DEVI u16 f2bf(float f) {
  unsigned u = __float_as_uint(f);
  u += 0x7fffu + ((u >> 16) & 1u);     // round-to-nearest-even
  return (u16)(u >> 16);
}
DEVI float bf2f(u16 h) { return __uint_as_float(((unsigned)h) << 16); }

// A fragment (16x32, bf16): lane = row (l&15); K elems e: (e<8?e:e+8) + half*8
DEVI bf16v load_a_frag(const u16* p, int lh) {
  const __bf16* b = reinterpret_cast<const __bf16*>(p) + lh * 8;
  bf8v lo = *reinterpret_cast<const bf8v*>(b);
  bf8v hi = *reinterpret_cast<const bf8v*>(b + 16);
  return __builtin_shufflevector(lo, hi, 0,1,2,3,4,5,6,7,8,9,10,11,12,13,14,15);
}
// B fragment (32x16, bf16) sourced from row-major X where B[k][n]=X[n][k]:
// lane = col (l&15); K elems e: (e<8?e:e+8) + half*8
DEVI bf16v load_b_frag(const u16* p, int lh) {
  const __bf16* b = reinterpret_cast<const __bf16*>(p) + lh * 8;
  bf8v lo = *reinterpret_cast<const bf8v*>(b);
  bf8v hi = *reinterpret_cast<const bf8v*>(b + 16);
  return __builtin_shufflevector(lo, hi, 0,1,2,3,4,5,6,7,8,9,10,11,12,13,14,15);
}
DEVI f8v wmma_bf(bf16v a, bf16v b, f8v c) {
  return __builtin_amdgcn_wmma_f32_16x16x32_bf16(false, a, false, b, (short)0, c,
                                                 false, false);
}

// ---------- setup kernels ----------
__global__ __launch_bounds__(256) void k_tr_src(const float* __restrict__ src,
                                                float* __restrict__ x) {
  int t = blockIdx.x * 256 + threadIdx.x;          // over NTOK*CDIM
  int i = t / CDIM, c = t % CDIM;
  x[t] = src[(size_t)c * NTOK + i];
}

__global__ __launch_bounds__(256) void k_conv1(const float* __restrict__ src,
                                               const float* __restrict__ w,
                                               const float* __restrict__ bias,
                                               float* __restrict__ out) {
  int t = blockIdx.x * 256 + threadIdx.x;          // over CMID*NTOK
  if (t >= CMID * NTOK) return;
  int co = t / (HH * WWI);
  int p  = t % (HH * WWI);
  int y = p / WWI, x = p % WWI;
  float acc = bias[co];
  const float* wp = w + (size_t)co * CDIM * 9;
  for (int ci = 0; ci < CDIM; ++ci) {
    const float* sp = src + (size_t)ci * HH * WWI;
    const float* wk = wp + ci * 9;
    #pragma unroll
    for (int dy = -1; dy <= 1; ++dy) {
      int yy = y + dy; if (yy < 0 || yy >= HH) continue;
      #pragma unroll
      for (int dx = -1; dx <= 1; ++dx) {
        int xx = x + dx; if (xx < 0 || xx >= WWI) continue;
        acc += sp[yy * WWI + xx] * wk[(dy + 1) * 3 + (dx + 1)];
      }
    }
  }
  out[t] = fmaxf(acc, 0.0f);                       // relu
}

__global__ __launch_bounds__(256) void k_conv2(const float* __restrict__ mid,
                                               const float* __restrict__ w,
                                               const float* __restrict__ bias,
                                               float* __restrict__ gp) {
  int t = blockIdx.x * 256 + threadIdx.x;          // over NHD*NTOK
  if (t >= NHD * NTOK) return;
  int h = t / (HH * WWI);
  int p = t % (HH * WWI);
  int y = p / WWI, x = p % WWI;
  float acc = bias[h];
  const float* wp = w + (size_t)h * CMID * 9;
  for (int ci = 0; ci < CMID; ++ci) {
    const float* sp = mid + (size_t)ci * HH * WWI;
    const float* wk = wp + ci * 9;
    #pragma unroll
    for (int dy = -1; dy <= 1; ++dy) {
      int yy = y + dy; if (yy < 0 || yy >= HH) continue;
      #pragma unroll
      for (int dx = -1; dx <= 1; ++dx) {
        int xx = x + dx; if (xx < 0 || xx >= WWI) continue;
        acc += sp[yy * WWI + xx] * wk[(dy + 1) * 3 + (dx + 1)];
      }
    }
  }
  gp[(size_t)p * NHD + h] = 1.0f / (1.0f + __expf(-acc));   // sigmoid, [n,nh]
}

__global__ __launch_bounds__(256) void k_f32_to_bf16(const float* __restrict__ in,
                                                     u16* __restrict__ out, int n) {
  int t = blockIdx.x * 256 + threadIdx.x;
  if (t < n) out[t] = f2bf(in[t]);
}

__global__ __launch_bounds__(256) void k_zero_i32(int* __restrict__ p, int n) {
  int t = blockIdx.x * 256 + threadIdx.x;
  if (t < n) p[t] = 0;
}

// split x into hi/lo bf16 halves + row sum of squares
__global__ __launch_bounds__(256) void k_split_sq(const float* __restrict__ x,
                                                  u16* __restrict__ xhi,
                                                  u16* __restrict__ xlo,
                                                  float* __restrict__ sq) {
  __shared__ float sm[256];
  int i = blockIdx.x, c = threadIdx.x;
  float v = x[(size_t)i * CDIM + c];
  u16 h = f2bf(v);
  xhi[(size_t)i * CDIM + c] = h;
  xlo[(size_t)i * CDIM + c] = f2bf(v - bf2f(h));
  sm[c] = v * v;
  __syncthreads();
  for (int s = 128; s > 0; s >>= 1) {
    if (c < s) sm[c] += sm[c + s];
    __syncthreads();
  }
  if (c == 0) sq[i] = sm[0];
}

// ---------- Gram matrix chunk via split-bf16 WMMA, 32x32 tile per wave ----------
__global__ __launch_bounds__(64) void k_gram_chunk(const u16* __restrict__ xhi,
                                                   const u16* __restrict__ xlo,
                                                   float* __restrict__ G, int q0) {
  int lane = threadIdx.x & 31, wv = threadIdx.x >> 5;
  int ln = lane & 15, lh = lane >> 4;
  int rb = blockIdx.y * 32;                 // row tile base within chunk (0..48)*32
  int cb = (blockIdx.x * 2 + wv) * 32;      // col base (0..97)*32
  const u16* ah0p = xhi + (size_t)(q0 + rb + ln) * CDIM;
  const u16* ah1p = xhi + (size_t)(q0 + rb + 16 + ln) * CDIM;
  const u16* al0p = xlo + (size_t)(q0 + rb + ln) * CDIM;
  const u16* al1p = xlo + (size_t)(q0 + rb + 16 + ln) * CDIM;
  const u16* bh0p = xhi + (size_t)(cb + ln) * CDIM;
  const u16* bh1p = xhi + (size_t)(cb + 16 + ln) * CDIM;
  const u16* bl0p = xlo + (size_t)(cb + ln) * CDIM;
  const u16* bl1p = xlo + (size_t)(cb + 16 + ln) * CDIM;
  f8v a00 = {0,0,0,0,0,0,0,0}, a01 = {0,0,0,0,0,0,0,0};
  f8v a10 = {0,0,0,0,0,0,0,0}, a11 = {0,0,0,0,0,0,0,0};
  for (int k0 = 0; k0 < CDIM; k0 += 32) {
    bf16v ah0 = load_a_frag(ah0p + k0, lh);
    bf16v ah1 = load_a_frag(ah1p + k0, lh);
    bf16v al0 = load_a_frag(al0p + k0, lh);
    bf16v al1 = load_a_frag(al1p + k0, lh);
    bf16v bh0 = load_b_frag(bh0p + k0, lh);
    bf16v bh1 = load_b_frag(bh1p + k0, lh);
    bf16v bl0 = load_b_frag(bl0p + k0, lh);
    bf16v bl1 = load_b_frag(bl1p + k0, lh);
    a00 = wmma_bf(ah0, bh0, a00); a00 = wmma_bf(ah0, bl0, a00); a00 = wmma_bf(al0, bh0, a00);
    a01 = wmma_bf(ah0, bh1, a01); a01 = wmma_bf(ah0, bl1, a01); a01 = wmma_bf(al0, bh1, a01);
    a10 = wmma_bf(ah1, bh0, a10); a10 = wmma_bf(ah1, bl0, a10); a10 = wmma_bf(al1, bh0, a10);
    a11 = wmma_bf(ah1, bh1, a11); a11 = wmma_bf(ah1, bl1, a11); a11 = wmma_bf(al1, bh1, a11);
  }
  #pragma unroll
  for (int r = 0; r < 8; ++r) {
    size_t r0 = (size_t)(rb + r + lh * 8) * NTOK;
    size_t r1 = (size_t)(rb + 16 + r + lh * 8) * NTOK;
    G[r0 + cb + ln]      = a00[r];
    G[r0 + cb + 16 + ln] = a01[r];
    G[r1 + cb + ln]      = a10[r];
    G[r1 + cb + 16 + ln] = a11[r];
  }
}

// ---------- top-k (16 smallest d2 per row) + in-degree scatter ----------
__global__ __launch_bounds__(256) void k_topk(const float* __restrict__ G,
                                              const float* __restrict__ sq,
                                              int* __restrict__ nearc, int q0) {
  __shared__ float rowv[NTOK];
  __shared__ float sval[256];
  __shared__ int   sidx[256];
  int tid = threadIdx.x;
  int rl = blockIdx.x;                  // row within chunk
  const float* grow = G + (size_t)rl * NTOK;
  for (int j = tid; j < NTOK; j += 256)
    rowv[j] = 2.0f * grow[j] - sq[j];   // ordering key of -d2 (row-affine)
  __syncthreads();
  for (int it = 0; it < TOPK_; ++it) {
    float bv = -3.4e38f; int bi = 0x7fffffff;
    for (int j = tid; j < NTOK; j += 256) {
      float v = rowv[j];
      if (v > bv) { bv = v; bi = j; }   // ascending j => lowest index on tie
    }
    sval[tid] = bv; sidx[tid] = bi;
    __syncthreads();
    for (int s = 128; s > 0; s >>= 1) {
      if (tid < s) {
        float v2 = sval[tid + s]; int i2 = sidx[tid + s];
        if (v2 > sval[tid] || (v2 == sval[tid] && i2 < sidx[tid])) {
          sval[tid] = v2; sidx[tid] = i2;
        }
      }
      __syncthreads();
    }
    if (tid == 0) {
      int best = sidx[0];
      atomicAdd(&nearc[best], 1);
      rowv[best] = -3.4e38f;            // mask out
    }
    __syncthreads();
  }
  (void)q0;
}

__global__ __launch_bounds__(256) void k_near_max(const int* __restrict__ nearc,
                                                  int* __restrict__ outmax) {
  __shared__ int sm[256];
  int tid = threadIdx.x, v = 0;
  for (int j = tid; j < NTOK; j += 256) v = max(v, nearc[j]);
  sm[tid] = v;
  __syncthreads();
  for (int s = 128; s > 0; s >>= 1) {
    if (tid < s) sm[tid] = max(sm[tid], sm[tid + s]);
    __syncthreads();
  }
  if (tid == 0) *outmax = sm[0];
}

__global__ __launch_bounds__(256) void k_build_qkin(const float* __restrict__ x,
                                                    const int* __restrict__ nearc,
                                                    const int* __restrict__ nearmax,
                                                    const float* __restrict__ nemb,
                                                    u16* __restrict__ qkin) {
  int i = blockIdx.x, c = threadIdx.x;
  float mx = (float)(*nearmax);
  int ni = (int)((float)nearc[i] / mx * (float)(USEN - 1));
  ni = min(max(ni, 0), USEN - 1);
  float v = x[(size_t)i * CDIM + c] + nemb[(size_t)ni * CDIM + c];
  qkin[(size_t)i * CDIM + c] = f2bf(v);
}

// ---------- bf16 WMMA GEMM: out = A[M,K] @ W[N,K]^T + bias, 32x32/wave ----------
__global__ __launch_bounds__(128) void k_gemm_bf16(const u16* __restrict__ A,
                                                   const u16* __restrict__ W,
                                                   const float* __restrict__ bias,
                                                   float* __restrict__ outF,
                                                   u16* __restrict__ outB,
                                                   int M, int K, int Nout, int relu) {
  int lane = threadIdx.x & 31, wv = threadIdx.x >> 5;
  int ln = lane & 15, lh = lane >> 4;
  int nb = (blockIdx.x * 4 + wv) * 32;
  int mb = blockIdx.y * 32;
  const u16* ap0 = A + (size_t)(mb + ln) * K;
  const u16* ap1 = A + (size_t)(mb + 16 + ln) * K;
  const u16* wp0 = W + (size_t)(nb + ln) * K;
  const u16* wp1 = W + (size_t)(nb + 16 + ln) * K;
  f8v a00 = {0,0,0,0,0,0,0,0}, a01 = {0,0,0,0,0,0,0,0};
  f8v a10 = {0,0,0,0,0,0,0,0}, a11 = {0,0,0,0,0,0,0,0};
  for (int k0 = 0; k0 < K; k0 += 32) {
    if (k0 + 32 < K) {                   // gfx1250 global_prefetch_b8 path
      __builtin_prefetch(ap0 + k0 + 32, 0, 1);
      __builtin_prefetch(wp0 + k0 + 32, 0, 1);
    }
    bf16v af0 = load_a_frag(ap0 + k0, lh);
    bf16v af1 = load_a_frag(ap1 + k0, lh);
    bf16v bf0 = load_b_frag(wp0 + k0, lh);
    bf16v bf1 = load_b_frag(wp1 + k0, lh);
    a00 = wmma_bf(af0, bf0, a00);
    a01 = wmma_bf(af0, bf1, a01);
    a10 = wmma_bf(af1, bf0, a10);
    a11 = wmma_bf(af1, bf1, a11);
  }
  float bv0 = bias[nb + ln];
  float bv1 = bias[nb + 16 + ln];
  #pragma unroll
  for (int r = 0; r < 8; ++r) {
    float v00 = a00[r] + bv0, v01 = a01[r] + bv1;
    float v10 = a10[r] + bv0, v11 = a11[r] + bv1;
    if (relu) {
      v00 = fmaxf(v00, 0.0f); v01 = fmaxf(v01, 0.0f);
      v10 = fmaxf(v10, 0.0f); v11 = fmaxf(v11, 0.0f);
    }
    size_t o0 = (size_t)(mb + r + lh * 8) * Nout + nb + ln;
    size_t o1 = (size_t)(mb + 16 + r + lh * 8) * Nout + nb + ln;
    if (outF) {
      outF[o0] = v00; outF[o0 + 16] = v01;
      outF[o1] = v10; outF[o1 + 16] = v11;
    }
    if (outB) {
      outB[o0] = f2bf(v00); outB[o0 + 16] = f2bf(v01);
      outB[o1] = f2bf(v10); outB[o1 + 16] = f2bf(v11);
    }
  }
  (void)M;
}

__global__ __launch_bounds__(256) void k_vtr(const u16* __restrict__ vbf,
                                             u16* __restrict__ vT) {
  int t = blockIdx.x * 256 + threadIdx.x;          // over NTOK*CDIM
  int j = t / CDIM, c = t % CDIM;
  vT[(size_t)c * NTOK + j] = vbf[t];
}

// ---------- flash attention, one wave per (head, 16-query tile) ----------
__global__ __launch_bounds__(32) void k_flash(const u16* __restrict__ qbf,
                                              const u16* __restrict__ kbf,
                                              const u16* __restrict__ vT,
                                              const float* __restrict__ gp,
                                              u16* __restrict__ obf, float scale) {
  __shared__ __align__(16) u16 Pb[16 * 32];
  int h = blockIdx.x;
  int qbase = blockIdx.y * 16;
  int lane = threadIdx.x;
  int ln = lane & 15, lh = lane >> 4;

  bf16v qa = load_a_frag(qbf + (size_t)(qbase + ln) * CDIM + h * DH, lh);

  float gpi[8];
  #pragma unroll
  for (int r = 0; r < 8; ++r)
    gpi[r] = gp[(size_t)(qbase + r + lh * 8) * NHD + h];

  f8v o0 = {0, 0, 0, 0, 0, 0, 0, 0};
  f8v o1 = {0, 0, 0, 0, 0, 0, 0, 0};
  float mrow[8], lrow[8];
  #pragma unroll
  for (int r = 0; r < 8; ++r) { mrow[r] = -3.0e38f; lrow[r] = 0.0f; }

  for (int kb = 0; kb < NTOK / 32; ++kb) {
    int k0 = kb * 32;
    f8v s0 = {0, 0, 0, 0, 0, 0, 0, 0};
    f8v s1 = {0, 0, 0, 0, 0, 0, 0, 0};
    s0 = wmma_bf(qa, load_b_frag(kbf + (size_t)(k0 + ln) * CDIM + h * DH, lh), s0);
    s1 = wmma_bf(qa, load_b_frag(kbf + (size_t)(k0 + 16 + ln) * CDIM + h * DH, lh), s1);
    float gpj0 = gp[(size_t)(k0 + ln) * NHD + h];
    float gpj1 = gp[(size_t)(k0 + 16 + ln) * NHD + h];
    #pragma unroll
    for (int r = 0; r < 8; ++r) {
      float a0 = s0[r] * scale + fabsf(gpi[r] - gpj0);
      float a1 = s1[r] * scale + fabsf(gpi[r] - gpj1);
      float t = fmaxf(a0, a1);
      t = fmaxf(t, __shfl_xor(t, 1, 32));
      t = fmaxf(t, __shfl_xor(t, 2, 32));
      t = fmaxf(t, __shfl_xor(t, 4, 32));
      t = fmaxf(t, __shfl_xor(t, 8, 32));
      float mn = fmaxf(mrow[r], t);
      float al = __expf(mrow[r] - mn);
      float p0 = __expf(a0 - mn);
      float p1 = __expf(a1 - mn);
      float ps = p0 + p1;
      ps += __shfl_xor(ps, 1, 32);
      ps += __shfl_xor(ps, 2, 32);
      ps += __shfl_xor(ps, 4, 32);
      ps += __shfl_xor(ps, 8, 32);
      lrow[r] = lrow[r] * al + ps;
      mrow[r] = mn;
      o0[r] *= al;
      o1[r] *= al;
      Pb[(r + lh * 8) * 32 + ln]      = f2bf(p0);
      Pb[(r + lh * 8) * 32 + 16 + ln] = f2bf(p1);
    }
    __syncthreads();
    bf16v pa  = load_a_frag(&Pb[ln * 32], lh);
    bf16v vb0 = load_b_frag(vT + (size_t)(h * DH + ln) * NTOK + k0, lh);
    bf16v vb1 = load_b_frag(vT + (size_t)(h * DH + 16 + ln) * NTOK + k0, lh);
    o0 = wmma_bf(pa, vb0, o0);
    o1 = wmma_bf(pa, vb1, o1);
    __syncthreads();
  }
  #pragma unroll
  for (int r = 0; r < 8; ++r) {
    float inv = 1.0f / lrow[r];
    size_t m = (size_t)(qbase + r + lh * 8);
    obf[m * CDIM + h * DH + ln]      = f2bf(o0[r] * inv);
    obf[m * CDIM + h * DH + 16 + ln] = f2bf(o1[r] * inv);
  }
}

// ---------- residual + layernorm (optionally transpose-out for final) ----------
__global__ __launch_bounds__(256) void k_add_ln(const float* __restrict__ xin,
                                                const float* __restrict__ delta,
                                                const float* __restrict__ g,
                                                const float* __restrict__ b,
                                                float* __restrict__ outF,
                                                u16* __restrict__ outB,
                                                float* __restrict__ outTr) {
  __shared__ float sm[256];
  __shared__ float stat[2];
  int i = blockIdx.x, c = threadIdx.x;
  float v = xin[(size_t)i * CDIM + c];
  if (delta) v += delta[(size_t)i * CDIM + c];
  sm[c] = v;
  __syncthreads();
  for (int s = 128; s > 0; s >>= 1) { if (c < s) sm[c] += sm[c + s]; __syncthreads(); }
  if (c == 0) stat[0] = sm[0] * (1.0f / CDIM);
  __syncthreads();
  float mean = stat[0];
  float d = v - mean;
  sm[c] = d * d;
  __syncthreads();
  for (int s = 128; s > 0; s >>= 1) { if (c < s) sm[c] += sm[c + s]; __syncthreads(); }
  if (c == 0) stat[1] = sm[0] * (1.0f / CDIM);
  __syncthreads();
  float y = d * rsqrtf(stat[1] + 1e-5f) * g[c] + b[c];
  if (outF) outF[(size_t)i * CDIM + c] = y;
  if (outB) outB[(size_t)i * CDIM + c] = f2bf(y);
  if (outTr) outTr[(size_t)c * NTOK + i] = y;
}

// ---------- host orchestration ----------
extern "C" void kernel_launch(void* const* d_in, const int* in_sizes, int n_in,
                              void* d_out, int out_size, void* d_ws, size_t ws_size,
                              hipStream_t stream) {
  (void)in_sizes; (void)n_in; (void)out_size; (void)ws_size;
  const float* src  = (const float*)d_in[0];
  const float* c1w  = (const float*)d_in[1];
  const float* c1b  = (const float*)d_in[2];
  const float* c2w  = (const float*)d_in[3];
  const float* c2b  = (const float*)d_in[4];
  const float* nemb = (const float*)d_in[5];
  const float* Wq = (const float*)d_in[6];  const float* bq = (const float*)d_in[7];
  const float* Wk = (const float*)d_in[8];  const float* bk = (const float*)d_in[9];
  const float* Wv = (const float*)d_in[10]; const float* bv = (const float*)d_in[11];
  const float* Wo = (const float*)d_in[12]; const float* bo = (const float*)d_in[13];
  const float* W1 = (const float*)d_in[14]; const float* b1 = (const float*)d_in[15];
  const float* W2 = (const float*)d_in[16]; const float* b2 = (const float*)d_in[17];
  const float* ln1g = (const float*)d_in[18]; const float* ln1b = (const float*)d_in[19];
  const float* ln2g = (const float*)d_in[20]; const float* ln2b = (const float*)d_in[21];
  const float* ng = (const float*)d_in[22];  const float* nbb = (const float*)d_in[23];
  float* out = (float*)d_out;

  char* base = (char*)d_ws;
  size_t off = 0;
  auto alloc = [&](size_t bytes) -> void* {
    void* p = base + off;
    off += (bytes + 255) & ~(size_t)255;
    return p;
  };
  float* x    = (float*)alloc((size_t)NTOK * CDIM * 4);
  float* x2   = (float*)alloc((size_t)NTOK * CDIM * 4);
  float* tmp1 = (float*)alloc((size_t)NTOK * CDIM * 4);
  float* mid  = (float*)alloc((size_t)CMID * NTOK * 4);
  float* gp   = (float*)alloc((size_t)NTOK * NHD * 4);
  float* sq   = (float*)alloc((size_t)NTOK * 4);
  int* nearc  = (int*)alloc((size_t)NTOK * 4);
  int* nearmx = (int*)alloc(256);
  u16* xhi  = (u16*)alloc((size_t)NTOK * CDIM * 2);
  u16* xlo  = (u16*)alloc((size_t)NTOK * CDIM * 2);
  u16* qkin = (u16*)alloc((size_t)NTOK * CDIM * 2);
  u16* qb   = (u16*)alloc((size_t)NTOK * CDIM * 2);
  u16* kbuf = (u16*)alloc((size_t)NTOK * CDIM * 2);
  u16* vbuf = (u16*)alloc((size_t)NTOK * CDIM * 2);
  u16* vT   = (u16*)alloc((size_t)NTOK * CDIM * 2);
  u16* attb = (u16*)alloc((size_t)NTOK * CDIM * 2);
  u16* x2b  = (u16*)alloc((size_t)NTOK * CDIM * 2);
  u16* f1b  = (u16*)alloc((size_t)NTOK * DFF_ * 2);
  u16* wqb  = (u16*)alloc((size_t)LAYERS * CDIM * CDIM * 2);
  u16* wkb  = (u16*)alloc((size_t)LAYERS * CDIM * CDIM * 2);
  u16* wvb  = (u16*)alloc((size_t)LAYERS * CDIM * CDIM * 2);
  u16* wob  = (u16*)alloc((size_t)LAYERS * CDIM * CDIM * 2);
  u16* w1b  = (u16*)alloc((size_t)LAYERS * DFF_ * CDIM * 2);
  u16* w2b  = (u16*)alloc((size_t)LAYERS * CDIM * DFF_ * 2);
  float* G  = (float*)alloc((size_t)NQC * NTOK * 4);

  const float scale = 0.17677669529663687f;  // 1/sqrt(32)

  // setup
  k_tr_src<<<NTOK * CDIM / 256, 256, 0, stream>>>(src, x);
  k_conv1<<<CMID * NTOK / 256, 256, 0, stream>>>(src, c1w, c1b, mid);
  k_conv2<<<NHD * NTOK / 256, 256, 0, stream>>>(mid, c2w, c2b, gp);
  int nqk = LAYERS * CDIM * CDIM, nff = LAYERS * DFF_ * CDIM;
  k_f32_to_bf16<<<nqk / 256, 256, 0, stream>>>(Wq, wqb, nqk);
  k_f32_to_bf16<<<nqk / 256, 256, 0, stream>>>(Wk, wkb, nqk);
  k_f32_to_bf16<<<nqk / 256, 256, 0, stream>>>(Wv, wvb, nqk);
  k_f32_to_bf16<<<nqk / 256, 256, 0, stream>>>(Wo, wob, nqk);
  k_f32_to_bf16<<<nff / 256, 256, 0, stream>>>(W1, w1b, nff);
  k_f32_to_bf16<<<nff / 256, 256, 0, stream>>>(W2, w2b, nff);

  for (int l = 0; l < LAYERS; ++l) {
    // kNN graph + degree embedding
    k_split_sq<<<NTOK, 256, 0, stream>>>(x, xhi, xlo, sq);
    k_zero_i32<<<(NTOK + 255) / 256, 256, 0, stream>>>(nearc, NTOK);
    for (int ch = 0; ch < 2; ++ch) {
      int q0 = ch * NQC;
      k_gram_chunk<<<dim3(49, NQC / 32), 64, 0, stream>>>(xhi, xlo, G, q0);
      k_topk<<<NQC, 256, 0, stream>>>(G, sq, nearc, q0);
    }
    k_near_max<<<1, 256, 0, stream>>>(nearc, nearmx);
    k_build_qkin<<<NTOK, 256, 0, stream>>>(x, nearc, nearmx, nemb, qkin);

    // attention
    k_gemm_bf16<<<dim3(CDIM / 128, NTOK / 32), 128, 0, stream>>>(
        qkin, wqb + (size_t)l * CDIM * CDIM, bq + l * CDIM, nullptr, qb,
        NTOK, CDIM, CDIM, 0);
    k_gemm_bf16<<<dim3(CDIM / 128, NTOK / 32), 128, 0, stream>>>(
        qkin, wkb + (size_t)l * CDIM * CDIM, bk + l * CDIM, nullptr, kbuf,
        NTOK, CDIM, CDIM, 0);
    k_gemm_bf16<<<dim3(CDIM / 128, NTOK / 32), 128, 0, stream>>>(
        xhi, wvb + (size_t)l * CDIM * CDIM, bv + l * CDIM, nullptr, vbuf,
        NTOK, CDIM, CDIM, 0);
    k_vtr<<<NTOK * CDIM / 256, 256, 0, stream>>>(vbuf, vT);
    k_flash<<<dim3(NHD, NTOK / 16), 32, 0, stream>>>(qb, kbuf, vT, gp, attb, scale);
    k_gemm_bf16<<<dim3(CDIM / 128, NTOK / 32), 128, 0, stream>>>(
        attb, wob + (size_t)l * CDIM * CDIM, bo + l * CDIM, tmp1, nullptr,
        NTOK, CDIM, CDIM, 0);
    k_add_ln<<<NTOK, 256, 0, stream>>>(x, tmp1, ln1g + l * CDIM, ln1b + l * CDIM,
                                       x2, x2b, nullptr);

    // FFN
    k_gemm_bf16<<<dim3(DFF_ / 128, NTOK / 32), 128, 0, stream>>>(
        x2b, w1b + (size_t)l * DFF_ * CDIM, b1 + l * DFF_, nullptr, f1b,
        NTOK, CDIM, DFF_, 1);
    k_gemm_bf16<<<dim3(CDIM / 128, NTOK / 32), 128, 0, stream>>>(
        f1b, w2b + (size_t)l * CDIM * DFF_, b2 + l * CDIM, tmp1, nullptr,
        NTOK, DFF_, CDIM, 0);
    k_add_ln<<<NTOK, 256, 0, stream>>>(x2, tmp1, ln2g + l * CDIM, ln2b + l * CDIM,
                                       x, nullptr, nullptr);
  }

  // final norm + transpose to [1,C,H,W]
  k_add_ln<<<NTOK, 256, 0, stream>>>(x, nullptr, ng, nbb, nullptr, nullptr, out);
}
